// Conv2d_dwa_44770739093947
// MI455X (gfx1250) — compile-verified, matrix operand order checked by reading, not run on hardware
//
#include <hip/hip_runtime.h>
#include <hip/hip_bf16.h>

// ---------------------------------------------------------------------------
// Problem: per-sample masked 3x3 conv, B=32, FIN=FOUT=256, 56x56, pad 1.
// Implicit GEMM per sample: M=FOUT=256, N=HW=3136, K=FIN*9=2304.
// ~118 GFLOP, ~283 MB HBM -> roughly balanced at 23.3 TB/s; bf16 WMMA path.
// Main kernel: double-buffered pipeline, A tiles via TDM (tensor_load_to_lds),
// B tiles global->VGPR->LDS with the ds stores pushed past the WMMA block.
// ---------------------------------------------------------------------------
#define BATCH 32
#define FIN   256
#define FOUT  256
#define HH    56
#define WW    56
#define HP    58              // padded spatial (56 + 2)
#define KTAPS 9               // 3x3
#define KROW  2304            // FIN * 9
#define NSTAGE 24             // 3 kh * 8 fin-chunks

#define XP_ELEMS  ((size_t)BATCH * FIN * HP * HP)
#define XP_BYTES  (XP_ELEMS * 2)

typedef __bf16 bf16_t;
typedef bf16_t v16bf __attribute__((ext_vector_type(16)));
typedef float  v8f   __attribute__((ext_vector_type(8)));

// TDM descriptor vector types (clang-23 6-arg form, probe-confirmed)
typedef unsigned int u32x4 __attribute__((ext_vector_type(4)));
typedef int          i32x8 __attribute__((ext_vector_type(8)));
typedef int          i32x4 __attribute__((ext_vector_type(4)));

#if __has_builtin(__builtin_amdgcn_tensor_load_to_lds)
#define HAVE_TDM 1
#else
#define HAVE_TDM 0
#endif

__device__ __forceinline__ void wait_tensorcnt0() {
#if __has_builtin(__builtin_amdgcn_s_wait_tensorcnt)
  __builtin_amdgcn_s_wait_tensorcnt(0);
#else
  asm volatile("s_wait_tensorcnt 0" ::: "memory");
#endif
}

__device__ __forceinline__ unsigned short f32_bf16(float f) {
  unsigned int u = __builtin_bit_cast(unsigned int, f);
  unsigned int lsb = (u >> 16) & 1u;
  u += 0x7fffu + lsb;                       // round-to-nearest-even
  return (unsigned short)(u >> 16);
}

// ---------------------------------------------------------------------------
// Kernel 1: zero-padded bf16 copy of x:  xp[b][fin][58][58]
// ---------------------------------------------------------------------------
__global__ __launch_bounds__(256) void pad_x(const float* __restrict__ x,
                                             unsigned short* __restrict__ xp) {
  size_t idx = (size_t)blockIdx.x * 256 + threadIdx.x;
  if (idx >= XP_ELEMS) return;
  int wp = (int)(idx % HP);
  size_t t = idx / HP;
  int hp = (int)(t % HP);
  size_t c = t / HP;                        // c = b*FIN + fin
  unsigned short v = 0;
  if (hp >= 1 && hp <= HH && wp >= 1 && wp <= WW) {
    v = f32_bf16(x[(c * HH + (size_t)(hp - 1)) * WW + (wp - 1)]);
  }
  xp[idx] = v;
}

// ---------------------------------------------------------------------------
// Kernel 2: masked weights, bf16, tap-major:  wt[b][fout][kk][fin]
// m and weight read fully coalesced; (fin,kk)->(kk,fin) transpose via LDS
// (read stride 9 floats, gcd(9,64)=1 -> bank-conflict free).
// ---------------------------------------------------------------------------
__global__ __launch_bounds__(256) void mask_w(const float* __restrict__ m,
                                              const float* __restrict__ w,
                                              unsigned short* __restrict__ wt) {
  __shared__ float p[KROW];
  int fout = blockIdx.x;
  int b    = blockIdx.y;
  int tid  = threadIdx.x;
  const float* mb = m + ((size_t)b * FOUT + fout) * KROW;
  const float* wb = w + (size_t)fout * KROW;
  for (int i = tid; i < KROW; i += 256) p[i] = mb[i] * wb[i];
  __syncthreads();
  unsigned short* o = wt + ((size_t)b * FOUT + fout) * KROW;
  for (int i = tid; i < KROW; i += 256) {
    int kk  = i >> 8;
    int fin = i & 255;
    o[i] = f32_bf16(p[fin * KTAPS + kk]);
  }
}

// ---------------------------------------------------------------------------
// Kernel 3: implicit-GEMM conv via v_wmma_f32_16x16x32_bf16, double-buffered.
// Block = (b, 128 fouts, 2 output rows). 8 waves: wave (wm,wn) owns
// M tile [wm*32,+32) x cols [wn*32,+32) of BOTH rows -> 8x v8f acc.
// Stage t = (kh, fin-chunk): A = 3 kw taps of 128x32 in ONE 3-D TDM op
// (LDS pad: +4 dwords every 16 dwords -> 80 B rows); B = 2 rows x 58 cols x
// 32 fin staged N-major so each kw tap is a pure LDS row shift.
// ---------------------------------------------------------------------------
#define LDSTRIDE 40                         // 32 + 8 pad halves -> 80 B rows

__device__ __forceinline__ v16bf load_frag(const unsigned short* row, int k0) {
  // ISA 16-bit fragment layout: lane holds K in {k0..k0+7, k0+16..k0+23}
  union { uint4 u[2]; v16bf v; } f;
  f.u[0] = *(const uint4*)(row + k0);
  f.u[1] = *(const uint4*)(row + k0 + 16);
  return f.v;
}

__global__ __launch_bounds__(256) void conv_wmma(
    const unsigned short* __restrict__ xp,   // bf16 [B][FIN][58][58]
    const unsigned short* __restrict__ wt,   // bf16 [B][FOUT][9][FIN]
    const float* __restrict__ bias,
    float* __restrict__ out) {
  __shared__ unsigned short As[2][128][3][LDSTRIDE]
      __attribute__((aligned(16)));
  __shared__ unsigned short Bs[2][2][66][LDSTRIDE]
      __attribute__((aligned(16)));

  const int b         = blockIdx.z;
  const int fout_base = blockIdx.y * 128;
  const int h0        = blockIdx.x * 2;      // two output rows per block
  const int tid       = threadIdx.x;
  const int wave      = tid >> 5;
  const int lane      = tid & 31;
  const int wm        = wave >> 1;           // 0..3 -> M offset wm*32
  const int wn        = wave & 1;            // 0..1 -> col offset wn*32

  v8f acc[2][2][2] = {};                     // [mi][hrow][nj]

  const int k0   = (lane < 16) ? 0 : 8;
  const int lrow = lane & 15;

  // B-staging geometry (fixed per thread)
  const int bf  = tid >> 3;                  // 0..31 fin_local
  const int bn0 = (tid & 7) * 8;             // 8 padded columns per thread

  // ---- stage-A issue: one 3-D TDM descriptor (wave 0 only; EXEC ignored
  //      by TDM so it must be branch-guarded) ----
  auto issueA = [&](int t, int s) {
#if HAVE_TDM
    if (wave == 0) {
      const int kh = t >> 3, fin0 = (t & 7) << 5;
      const unsigned lds_base = (unsigned)(size_t)&As[s][0][0][0];
      const unsigned long long ga = (unsigned long long)(size_t)(
          wt + ((((size_t)b * FOUT + fout_base) * KTAPS + kh * 3) * FIN +
                fin0));
      u32x4 g0;
      g0[0] = 1u;                               // count=1, user descriptor
      g0[1] = lds_base;                         // lds_addr
      g0[2] = (unsigned)ga;                     // global_addr[31:0]
      g0[3] = ((unsigned)(ga >> 32) & 0x01FFFFFFu) | (2u << 30);  // type=2
      i32x8 g1;
      // data_size=2B | pad_enable | pad_interval=16dw | pad_amount=4dw
      g1[0] = (int)(0x00010000u | (1u << 20) | (3u << 22) | (3u << 25));
      g1[1] = (int)(32u << 16);                 // tensor_dim0 = 32
      g1[2] = (int)(3u << 16);                  // tensor_dim1 = 3
      g1[3] = (int)(32u << 16);                 // tile_dim0 = 32
      g1[4] = (int)((128u << 16) | 3u);         // tile_dim1=3, tile_dim2=128
      g1[5] = 256;                              // dim0_stride (kw step)
      g1[6] = (int)(2304u << 16);               // dim1_stride lo (fout step)
      g1[7] = 0;                                // dim1_stride hi
      i32x4 g2;
      g2[0] = 128;                              // tensor_dim2 = 128
      g2[1] = 0; g2[2] = 0; g2[3] = 0;
      i32x4 g3 = {0, 0, 0, 0};
      i32x8 g4 = {0, 0, 0, 0, 0, 0, 0, 0};      // extra group (clang-23 ABI)
      __builtin_amdgcn_tensor_load_to_lds(g0, g1, g2, g3, g4, 0);
    }
#else
    {
      const int kh = t >> 3, fin0 = (t & 7) << 5;
      const int r = tid >> 1, half = tid & 1;
      const unsigned short* srcA =
          wt + ((((size_t)b * FOUT + fout_base + r) * KTAPS + kh * 3) * FIN +
                fin0 + half * 16);
#pragma unroll
      for (int kw = 0; kw < 3; ++kw) {
        *(uint4*)&As[s][r][kw][half * 16] = *(const uint4*)(srcA + kw * FIN);
        *(uint4*)&As[s][r][kw][half * 16 + 8] =
            *(const uint4*)(srcA + kw * FIN + 8);
      }
    }
#endif
  };

  auto loadB = [&](int t, unsigned short regs[16]) {
    const int kh = t >> 3, fin0 = (t & 7) << 5;
    const unsigned short* base =
        xp + ((size_t)b * FIN + fin0 + bf) * (HP * HP);
#pragma unroll
    for (int r = 0; r < 2; ++r) {
      const unsigned short* src = base + (h0 + r + kh) * HP + bn0;
#pragma unroll
      for (int i = 0; i < 8; ++i)
        regs[r * 8 + i] = (bn0 + i < HP) ? src[i] : (unsigned short)0;
    }
  };

  auto storeB = [&](const unsigned short regs[16], int s) {
#pragma unroll
    for (int r = 0; r < 2; ++r)
#pragma unroll
      for (int i = 0; i < 8; ++i)
        Bs[s][r][bn0 + i][bf] = regs[r * 8 + i];
  };

  auto compute = [&](int s) {
#pragma unroll
    for (int kw = 0; kw < 3; ++kw) {
      v16bf afrag[2];
#pragma unroll
      for (int i = 0; i < 2; ++i)
        afrag[i] = load_frag(&As[s][wm * 32 + i * 16 + lrow][kw][0], k0);
#pragma unroll
      for (int r = 0; r < 2; ++r) {
#pragma unroll
        for (int j = 0; j < 2; ++j) {
          v16bf bfrag =
              load_frag(&Bs[s][r][wn * 32 + j * 16 + lrow + kw][0], k0);
#pragma unroll
          for (int i = 0; i < 2; ++i)
            acc[i][r][j] = __builtin_amdgcn_wmma_f32_16x16x32_bf16(
                false, afrag[i], false, bfrag, (short)0, acc[i][r][j], false,
                false);
        }
      }
    }
  };

  // ---- prologue: fill buffer 0 ----
  {
    unsigned short regs[16];
    issueA(0, 0);
    loadB(0, regs);
    storeB(regs, 0);
  }
  if (wave == 0) wait_tensorcnt0();
  __syncthreads();

  // ---- pipelined main loop: one barrier per stage ----
  for (int t = 0; t < NSTAGE; ++t) {
    const int s = t & 1;
    unsigned short regs[16];
    const bool more = (t + 1 < NSTAGE);
    if (more) {
      issueA(t + 1, s ^ 1);                 // async DMA overlaps WMMAs below
      loadB(t + 1, regs);                   // global loads overlap WMMAs too
    }
    compute(s);
    if (more) storeB(regs, s ^ 1);          // LDS stores after the math
    if (wave == 0) wait_tensorcnt0();
    __syncthreads();
  }

  // ---- epilogue: lane holds M = vr + (lane<16?0:8), N = lane&15 ----
  const int mhalf = (lane < 16) ? 0 : 8;
#pragma unroll
  for (int i = 0; i < 2; ++i) {
#pragma unroll
    for (int vr = 0; vr < 8; ++vr) {
      const int fout = fout_base + wm * 32 + i * 16 + vr + mhalf;
      const float bv = bias[fout];
#pragma unroll
      for (int r = 0; r < 2; ++r) {
        float* orow = out + (((size_t)b * FOUT + fout) * HH + (h0 + r)) * WW;
#pragma unroll
        for (int j = 0; j < 2; ++j) {
          const int wcol = wn * 32 + j * 16 + (lane & 15);
          if (wcol < WW) orow[wcol] = acc[i][r][j][vr] + bv;
        }
      }
    }
  }
}

// ---------------------------------------------------------------------------
extern "C" void kernel_launch(void* const* d_in, const int* in_sizes, int n_in,
                              void* d_out, int out_size, void* d_ws,
                              size_t ws_size, hipStream_t stream) {
  const float* x    = (const float*)d_in[0];
  const float* m    = (const float*)d_in[1];
  const float* wgt  = (const float*)d_in[2];
  const float* bias = (const float*)d_in[3];
  float* out = (float*)d_out;

  unsigned short* xp = (unsigned short*)d_ws;
  unsigned short* wt = (unsigned short*)((char*)d_ws + XP_BYTES);

  const int padBlocks = (int)((XP_ELEMS + 255) / 256);
  pad_x<<<padBlocks, 256, 0, stream>>>(x, xp);
  mask_w<<<dim3(FOUT, BATCH), 256, 0, stream>>>(m, wgt, wt);
  conv_wmma<<<dim3(HH / 2, 2, BATCH), 256, 0, stream>>>(xp, wt, bias, out);
}
